// PrimeKGDrugRepurposingGNN_56684978372941
// MI455X (gfx1250) — matrix operand order repaired and checked by reading, not compiled, not run on hardware
//
#include <hip/hip_runtime.h>
#include <math.h>

// ---------------------------------------------------------------------------
// Problem constants (match reference)
// ---------------------------------------------------------------------------
constexpr int Nn = 50000;   // nodes
constexpr int Hh = 512;     // hidden
constexpr int Dd = 256;     // out dim
constexpr int Ee = 200000;  // edges per relation
constexpr int Pp = 100000;  // pairs
constexpr int Rr = 3;       // relations
constexpr int K3 = 3 * Dd;  // 768

typedef __attribute__((ext_vector_type(4)))  __bf16 v4bf;
typedef __attribute__((ext_vector_type(8)))  __bf16 v8bf;
typedef __attribute__((ext_vector_type(16))) __bf16 v16bf;
typedef __attribute__((ext_vector_type(8)))  float  v8f;

__device__ __forceinline__ __bf16 f2bf(float f) {
    unsigned u = __builtin_bit_cast(unsigned, f);
    unsigned r = (u + 0x7FFFu + ((u >> 16) & 1u)) >> 16;
    return __builtin_bit_cast(__bf16, (unsigned short)r);
}

// Two f32 -> packed bf16 pair: round-half-up + one v_perm_b32 byte-pack.
__device__ __forceinline__ unsigned pack2bf(float a, float b) {
    unsigned ua = __builtin_bit_cast(unsigned, a) + 0x8000u;
    unsigned ub = __builtin_bit_cast(unsigned, b) + 0x8000u;
    // byte pool: [0..3] = ua, [4..7] = ub ; result = {ub[3],ub[2],ua[3],ua[2]}
    return __builtin_amdgcn_perm(ub, ua, 0x07060302);
}
__device__ __forceinline__ void store_pack4(void* dst, float4 v) {
    uint2 o;
    o.x = pack2bf(v.x, v.y);
    o.y = pack2bf(v.z, v.w);
    *(uint2*)dst = o;   // ds_store_b64
}

__device__ __forceinline__ v16bf join16(v8bf lo, v8bf hi) {
    return __builtin_shufflevector(lo, hi, 0,1,2,3,4,5,6,7,8,9,10,11,12,13,14,15);
}

// CDNA5 async global->LDS copy of 16 bytes (tracked by ASYNCcnt).
// LDS address = low 32 bits of the generic address (aperture mapping).
__device__ __forceinline__ void async_copy_b128(const void* gsrc, void* ldst) {
    unsigned lds_off = (unsigned)(unsigned long long)ldst;
    unsigned long long gaddr = (unsigned long long)gsrc;
    asm volatile("global_load_async_to_lds_b128 %0, %1, off"
                 :: "v"(lds_off), "v"(gaddr) : "memory");
}
__device__ __forceinline__ void wait_async0() {
    asm volatile("s_wait_asynccnt 0" ::: "memory");
}

// ---------------------------------------------------------------------------
// Weight pre-pass: out[n*K + k] = bf16(in[k*Nc + n])   (transpose + convert)
// ---------------------------------------------------------------------------
__global__ __launch_bounds__(256) void cvt_transpose_kernel(
    const float* __restrict__ in, __bf16* __restrict__ out, int K, int Nc) {
    int i = blockIdx.x * 256 + threadIdx.x;
    if (i < K * Nc) {
        int k = i / Nc, n = i - k * Nc;
        out[(size_t)n * K + k] = f2bf(in[i]);
    }
}

// ---------------------------------------------------------------------------
// x = node_emb + type_emb[type_ids]   (one block per node row)
// ---------------------------------------------------------------------------
__global__ __launch_bounds__(256) void encode_kernel(
    const float* __restrict__ node_emb, const float* __restrict__ type_emb,
    const int* __restrict__ type_ids, float* __restrict__ x) {
    int r = blockIdx.x;
    int t = type_ids[r];
    const float* a = node_emb + (size_t)r * Hh;
    const float* b = type_emb + (size_t)t * Hh;
    float* o = x + (size_t)r * Hh;
    for (int j = threadIdx.x; j < Hh; j += 256) o[j] = a[j] + b[j];
}

// ---------------------------------------------------------------------------
// msg[dst[e]] += x[src[e]]  (unweighted COO segment-sum via f32 atomics)
// ---------------------------------------------------------------------------
__global__ __launch_bounds__(256) void scatter_add_kernel(
    const float* __restrict__ x, const int* __restrict__ src,
    const int* __restrict__ dst, float* __restrict__ msg) {
    int e = blockIdx.x;
    int s = src[e], d = dst[e];
    const float* xs = x + (size_t)s * Hh;
    float* md = msg + (size_t)d * Hh;
    for (int j = threadIdx.x; j < Hh; j += 256)
        atomicAdd(md + j, xs[j]);
}

// ---------------------------------------------------------------------------
// WMMA GEMM: C[M,Ncols] (+)= A[M,K](f32 -> bf16 in LDS) @ Bt[Ncols,K](bf16, pre-T)
// Block tile 128x128, 8 waves (4x2), wave tile 32x64 (2x4 wmma), K step 32.
// Double-buffered LDS; B tiles staged with async global->LDS b128 copies.
// Requires: K % 32 == 0, Ncols % 128 == 0 (M may be ragged).
// ---------------------------------------------------------------------------
__global__ __launch_bounds__(256) void gemm_bf16_wmma_kernel(
    const float* __restrict__ A, const __bf16* __restrict__ Bt,
    float* __restrict__ C, int M, int K, int Ncols, int accumulate) {
    __shared__ __align__(16) __bf16 sA[2][128 * 32];   // [m][k]
    __shared__ __align__(16) __bf16 sBt[2][128 * 32];  // [n][k]

    const int tid   = threadIdx.x;
    const int lane  = tid & 31;
    const int l15   = lane & 15;
    const int lhalf = lane >> 4;           // 0 or 1
    const int wave  = tid >> 5;
    const int wm    = wave >> 1;           // 0..3  (rows of 32)
    const int wn    = wave & 1;            // 0..1  (cols of 64)
    const int mb    = blockIdx.y * 128;
    const int nb    = blockIdx.x * 128;

    // B staging: thread copies 16B x2: row n = it*64 + tid>>2, quad q = tid&3
    const int bq = tid & 3;

    auto stageB = [&](int buf, int k0) {
        #pragma unroll
        for (int it = 0; it < 2; ++it) {
            int n = it * 64 + (tid >> 2);
            async_copy_b128(Bt + (size_t)(nb + n) * K + k0 + bq * 8,
                            &sBt[buf][n * 32 + bq * 8]);
        }
    };
    auto stageA = [&](int buf, int k0) {
        #pragma unroll
        for (int it = 0; it < 4; ++it) {
            int l = (it * 256 + tid) * 4;       // linear = row*32 + col
            int r = l >> 5;
            int c = l & 31;
            int gm = mb + r;
            float4 v = make_float4(0.f, 0.f, 0.f, 0.f);
            if (gm < M) v = *(const float4*)(A + (size_t)gm * K + k0 + c);
            store_pack4(&sA[buf][l], v);
        }
    };

    v8f acc[2][4] = {};

    stageB(0, 0);
    stageA(0, 0);

    int p = 0;
    for (int k0 = 0; k0 < K; k0 += 32) {
        wait_async0();
        __syncthreads();
        if (k0 + 32 < K) {
            stageB(p ^ 1, k0 + 32);
            stageA(p ^ 1, k0 + 32);
        }

        v16bf afrag[2], bfrag[4];
        #pragma unroll
        for (int i = 0; i < 2; ++i) {
            const __bf16* rp = &sA[p][(wm * 32 + i * 16 + l15) * 32];
            v8bf lo = *(const v8bf*)(rp + lhalf * 8);        // K = kh..kh+7
            v8bf hi = *(const v8bf*)(rp + lhalf * 8 + 16);   // K = kh+16..kh+23
            afrag[i] = join16(lo, hi);
        }
        #pragma unroll
        for (int j = 0; j < 4; ++j) {
            const __bf16* cp = &sBt[p][(wn * 64 + j * 16 + l15) * 32];
            v8bf lo = *(const v8bf*)(cp + lhalf * 8);        // K = kh..kh+7
            v8bf hi = *(const v8bf*)(cp + lhalf * 8 + 16);   // K = kh+16..kh+23
            bfrag[j] = join16(lo, hi);
        }
        #pragma unroll
        for (int i = 0; i < 2; ++i)
            #pragma unroll
            for (int j = 0; j < 4; ++j)
                acc[i][j] = __builtin_amdgcn_wmma_f32_16x16x32_bf16(
                    false, afrag[i], false, bfrag[j], (short)0, acc[i][j],
                    false, false);
        p ^= 1;
    }

    // store D tile: lane (l15) = col, VGPR v -> row v + lhalf*8 within 16x16 tile
    #pragma unroll
    for (int i = 0; i < 2; ++i)
        #pragma unroll
        for (int j = 0; j < 4; ++j) {
            int n = nb + wn * 64 + j * 16 + l15;
            #pragma unroll
            for (int v = 0; v < 8; ++v) {
                int m = mb + wm * 32 + i * 16 + lhalf * 8 + v;
                if (m < M) {
                    size_t off = (size_t)m * Ncols + n;
                    C[off] = accumulate ? (C[off] + acc[i][j][v]) : acc[i][j][v];
                }
            }
        }
}

// ---------------------------------------------------------------------------
// x = relu(acc)   elementwise
// ---------------------------------------------------------------------------
__global__ __launch_bounds__(256) void relu_kernel(
    const float* __restrict__ in, float* __restrict__ out, int n) {
    int i = blockIdx.x * 256 + threadIdx.x;
    if (i < n) out[i] = fmaxf(in[i], 0.f);
}

// ---------------------------------------------------------------------------
// x += relu(layernorm(acc) * g + b)   one block per row (H=512)
// ---------------------------------------------------------------------------
__global__ __launch_bounds__(256) void ln_relu_res_kernel(
    const float* __restrict__ acc, const float* __restrict__ g,
    const float* __restrict__ b, float* __restrict__ x) {
    __shared__ float s1[256], s2[256];
    int r = blockIdx.x, tid = threadIdx.x;
    const float* row = acc + (size_t)r * Hh;
    float a0 = row[tid], a1 = row[tid + 256];
    s1[tid] = a0 + a1;
    s2[tid] = a0 * a0 + a1 * a1;
    __syncthreads();
    for (int st = 128; st > 0; st >>= 1) {
        if (tid < st) { s1[tid] += s1[tid + st]; s2[tid] += s2[tid + st]; }
        __syncthreads();
    }
    float mu  = s1[0] * (1.f / Hh);
    float var = s2[0] * (1.f / Hh) - mu * mu;
    float inv = rsqrtf(var + 1e-5f);
    float h0 = fmaxf((a0 - mu) * inv * g[tid] + b[tid], 0.f);
    float h1 = fmaxf((a1 - mu) * inv * g[tid + 256] + b[tid + 256], 0.f);
    x[(size_t)r * Hh + tid]       += h0;
    x[(size_t)r * Hh + tid + 256] += h1;
}

// ---------------------------------------------------------------------------
// Fused pair scorer:
//   feats = [z[s], z[d], z[s]*z[d]]  (built per 64x32 tile in LDS, bf16)
//   h1 = BN(relu(feats @ W1 + b1));  score = h1 @ w2 + b2
//   out = (score - relu(a)*log(sd) - relu(b)*log(dd)) / (sqrt(sd)*sqrt(dd)+1e-8)
// Block: 64 pairs, 8 waves (2x4), wave tile 32x64 (2x4 wmma), K=768.
// W1 pre-transposed [256][768] bf16; its tiles staged with async copies.
// ---------------------------------------------------------------------------
__global__ __launch_bounds__(256) void score_kernel(
    const float* __restrict__ z, const __bf16* __restrict__ w1t,
    const int* __restrict__ pairs, const int* __restrict__ degrees,
    const float* __restrict__ b1, const float* __restrict__ bn_g,
    const float* __restrict__ bn_b, const float* __restrict__ bn_mean,
    const float* __restrict__ bn_var, const float* __restrict__ w2,
    const float* __restrict__ b2p, const float* __restrict__ alphap,
    const float* __restrict__ betap, float* __restrict__ out) {
    __shared__ __align__(16) __bf16 sF[64 * 32];    // feats tile [p][k]
    __shared__ __align__(16) __bf16 sWt[256 * 32];  // W1^T tile [n][k]
    __shared__ int   sSrc[64], sDst[64];
    __shared__ float sRed[64];

    const int tid   = threadIdx.x;
    const int lane  = tid & 31;
    const int l15   = lane & 15;
    const int lhalf = lane >> 4;
    const int wave  = tid >> 5;
    const int wm    = wave >> 2;   // 0..1  (rows of 32)
    const int wn    = wave & 3;    // 0..3  (cols of 64)
    const int pb    = blockIdx.x * 64;

    if (tid < 64) {
        int p = pb + tid;
        sSrc[tid] = (p < Pp) ? pairs[p] : 0;
        sDst[tid] = (p < Pp) ? pairs[Pp + p] : 0;
        sRed[tid] = 0.f;
    }
    __syncthreads();

    v8f acc[2][4] = {};

    const int wq = tid & 3;        // quad within W1^T row
    for (int k0 = 0; k0 < K3; k0 += 32) {
        // async stage W1^T tile: 256 rows x 64B; 4 x b128 per thread
        #pragma unroll
        for (int it = 0; it < 4; ++it) {
            int n = it * 64 + (tid >> 2);
            async_copy_b128(w1t + (size_t)n * K3 + k0 + wq * 8,
                            &sWt[n * 32 + wq * 8]);
        }
        // VALU stage feats 64x32 (2048 elems, 2 iters x float4) -- overlaps async
        int seg = k0 >> 8;          // 0: src, 1: dst, 2: src*dst
        int kb  = k0 & 255;
        #pragma unroll
        for (int it = 0; it < 2; ++it) {
            int l  = (it * 256 + tid) * 4;   // linear = pr*32 + c
            int pr = l >> 5;
            int c  = l & 31;
            float4 v = make_float4(0.f, 0.f, 0.f, 0.f);
            if (pb + pr < Pp) {
                int s = sSrc[pr], d = sDst[pr];
                if (seg == 0) {
                    v = *(const float4*)(z + (size_t)s * Dd + kb + c);
                } else if (seg == 1) {
                    v = *(const float4*)(z + (size_t)d * Dd + kb + c);
                } else {
                    float4 va = *(const float4*)(z + (size_t)s * Dd + kb + c);
                    float4 vb = *(const float4*)(z + (size_t)d * Dd + kb + c);
                    v = make_float4(va.x * vb.x, va.y * vb.y, va.z * vb.z, va.w * vb.w);
                }
            }
            store_pack4(&sF[l], v);
        }
        wait_async0();
        __syncthreads();

        v16bf afrag[2], bfrag[4];
        #pragma unroll
        for (int i = 0; i < 2; ++i) {
            const __bf16* rp = &sF[(wm * 32 + i * 16 + l15) * 32];
            v8bf lo = *(const v8bf*)(rp + lhalf * 8);
            v8bf hi = *(const v8bf*)(rp + lhalf * 8 + 16);
            afrag[i] = join16(lo, hi);
        }
        #pragma unroll
        for (int j = 0; j < 4; ++j) {
            const __bf16* cp = &sWt[(wn * 64 + j * 16 + l15) * 32];
            v8bf lo = *(const v8bf*)(cp + lhalf * 8);
            v8bf hi = *(const v8bf*)(cp + lhalf * 8 + 16);
            bfrag[j] = join16(lo, hi);
        }
        #pragma unroll
        for (int i = 0; i < 2; ++i)
            #pragma unroll
            for (int j = 0; j < 4; ++j)
                acc[i][j] = __builtin_amdgcn_wmma_f32_16x16x32_bf16(
                    false, afrag[i], false, bfrag[j], (short)0, acc[i][j],
                    false, false);
        __syncthreads();
    }

    // epilogue: bias -> relu -> BN -> * w2, reduce per pair via LDS atomics
    #pragma unroll
    for (int j = 0; j < 4; ++j) {
        int n = wn * 64 + j * 16 + l15;
        float scale = bn_g[n] * rsqrtf(bn_var[n] + 1e-5f);
        float shift = bn_b[n] - bn_mean[n] * scale;
        float bias  = b1[n];
        float w2v   = w2[n];
        #pragma unroll
        for (int i = 0; i < 2; ++i) {
            #pragma unroll
            for (int v = 0; v < 8; ++v) {
                float h = fmaxf(acc[i][j][v] + bias, 0.f);
                h = h * scale + shift;
                int row = wm * 32 + i * 16 + lhalf * 8 + v;
                atomicAdd(&sRed[row], h * w2v);
            }
        }
    }
    __syncthreads();

    if (tid < 64) {
        int p = pb + tid;
        if (p < Pp) {
            float alpha = alphap[0], beta = betap[0];
            float s = sRed[tid] + b2p[0];
            int sdi = degrees[sSrc[tid]]; if (sdi < 1) sdi = 1;
            int ddi = degrees[sDst[tid]]; if (ddi < 1) ddi = 1;
            float sd = (float)sdi, dd = (float)ddi;
            float base = s - fmaxf(alpha, 0.f) * logf(sd)
                           - fmaxf(beta, 0.f) * logf(dd);
            out[p] = base / (sqrtf(sd) * sqrtf(dd) + 1e-8f);
        }
    }
}

// ---------------------------------------------------------------------------
// Host orchestration
// ---------------------------------------------------------------------------
extern "C" void kernel_launch(void* const* d_in, const int* in_sizes, int n_in,
                              void* d_out, int out_size, void* d_ws, size_t ws_size,
                              hipStream_t stream) {
    (void)in_sizes; (void)n_in; (void)out_size; (void)ws_size;

    // inputs in setup_inputs() order
    const int*   node_type = (const int*)d_in[0];
    const int*   srcs[Rr]  = {(const int*)d_in[1], (const int*)d_in[3], (const int*)d_in[5]};
    const int*   dsts[Rr]  = {(const int*)d_in[2], (const int*)d_in[4], (const int*)d_in[6]};
    const int*   pairs     = (const int*)d_in[7];
    const int*   degrees   = (const int*)d_in[8];
    const float* node_emb  = (const float*)d_in[9];
    const float* type_emb  = (const float*)d_in[10];
    const float* w_loop_in = (const float*)d_in[11];
    const float* w_rel_in  = (const float*)d_in[12];
    const float* w_loop_res= (const float*)d_in[13];
    const float* w_rel_res = (const float*)d_in[14];
    const float* ln_g      = (const float*)d_in[15];
    const float* ln_b      = (const float*)d_in[16];
    const float* w_loop_out= (const float*)d_in[17];
    const float* w_rel_out = (const float*)d_in[18];
    const float* lp_w1     = (const float*)d_in[19];
    const float* lp_b1     = (const float*)d_in[20];
    const float* bn_g      = (const float*)d_in[21];
    const float* bn_b      = (const float*)d_in[22];
    const float* bn_mean   = (const float*)d_in[23];
    const float* bn_var    = (const float*)d_in[24];
    const float* lp_w2     = (const float*)d_in[25];
    const float* lp_b2     = (const float*)d_in[26];
    const float* alpha     = (const float*)d_in[27];
    const float* beta      = (const float*)d_in[28];
    float* out             = (float*)d_out;

    // workspace carve-up (all bf16 weights stored TRANSPOSED: [Nc][K])
    float* x    = (float*)d_ws;          // N*H
    float* msg  = x   + (size_t)Nn * Hh; // N*H
    float* accb = msg + (size_t)Nn * Hh; // N*H
    float* z    = accb+ (size_t)Nn * Hh; // N*D
    __bf16* wb          = (__bf16*)(z + (size_t)Nn * Dd);
    __bf16* wb_loop_in  = wb;                                  // [H][H]
    __bf16* wb_rel_in   = wb_loop_in  + (size_t)Hh * Hh;       // R x [H][H]
    __bf16* wb_loop_res = wb_rel_in   + (size_t)Rr * Hh * Hh;
    __bf16* wb_rel_res  = wb_loop_res + (size_t)Hh * Hh;
    __bf16* wb_loop_out = wb_rel_res  + (size_t)Rr * Hh * Hh;  // [D][H]
    __bf16* wb_rel_out  = wb_loop_out + (size_t)Hh * Dd;       // R x [D][H]
    __bf16* wb_w1       = wb_rel_out  + (size_t)Rr * Hh * Dd;  // [256][768]

    auto cvtT = [&](const float* src, __bf16* dst, int K, int Nc) {
        int n = K * Nc;
        cvt_transpose_kernel<<<(n + 255) / 256, 256, 0, stream>>>(src, dst, K, Nc);
    };
    cvtT(w_loop_in,  wb_loop_in,  Hh, Hh);
    for (int r = 0; r < Rr; ++r)
        cvtT(w_rel_in  + (size_t)r * Hh * Hh, wb_rel_in  + (size_t)r * Hh * Hh, Hh, Hh);
    cvtT(w_loop_res, wb_loop_res, Hh, Hh);
    for (int r = 0; r < Rr; ++r)
        cvtT(w_rel_res + (size_t)r * Hh * Hh, wb_rel_res + (size_t)r * Hh * Hh, Hh, Hh);
    cvtT(w_loop_out, wb_loop_out, Hh, Dd);
    for (int r = 0; r < Rr; ++r)
        cvtT(w_rel_out + (size_t)r * Hh * Dd, wb_rel_out + (size_t)r * Hh * Dd, Hh, Dd);
    cvtT(lp_w1, wb_w1, K3, Dd);

    auto gemm = [&](const float* A, const __bf16* Bt, float* C,
                    int M, int K, int Nc, int accum) {
        dim3 grid(Nc / 128, (M + 127) / 128);
        gemm_bf16_wmma_kernel<<<grid, 256, 0, stream>>>(A, Bt, C, M, K, Nc, accum);
    };
    auto rgcn = [&](const float* xin, const __bf16* wl, const __bf16* wr,
                    float* cout, int Nc) {
        gemm(xin, wl, cout, Nn, Hh, Nc, 0);
        for (int r = 0; r < Rr; ++r) {
            hipMemsetAsync(msg, 0, (size_t)Nn * Hh * sizeof(float), stream);
            scatter_add_kernel<<<Ee, 256, 0, stream>>>(xin, srcs[r], dsts[r], msg);
            gemm(msg, wr + (size_t)r * Hh * Nc, cout, Nn, Hh, Nc, 1);
        }
    };

    // encode
    encode_kernel<<<Nn, 256, 0, stream>>>(node_emb, type_emb, node_type, x);

    // layer 1: x = relu(rgcn(x, in))
    rgcn(x, wb_loop_in, wb_rel_in, accb, Hh);
    relu_kernel<<<(Nn * Hh + 255) / 256, 256, 0, stream>>>(accb, x, Nn * Hh);

    // layer 2: x = x + relu(LN(rgcn(x, res)))
    rgcn(x, wb_loop_res, wb_rel_res, accb, Hh);
    ln_relu_res_kernel<<<Nn, 256, 0, stream>>>(accb, ln_g, ln_b, x);

    // layer 3: z = rgcn(x, out)   [N, D]
    rgcn(x, wb_loop_out, wb_rel_out, z, Dd);

    // fused link-prediction scorer
    score_kernel<<<(Pp + 63) / 64, 256, 0, stream>>>(
        z, wb_w1, pairs, degrees, lp_b1, bn_g, bn_b, bn_mean, bn_var,
        lp_w2, lp_b2, alpha, beta, out);
}
